// DeepClusteringModel_893353197862
// MI455X (gfx1250) — matrix operand order repaired and checked by reading, not compiled, not run on hardware
//
#include <hip/hip_runtime.h>
#include <hip/hip_bf16.h>

#define N_NODES 100000
#define E_EDGES 1600000
#define IN_DIM 128
#define HID 128
#define EMB 64
#define KCLU 100
#define LPAD 112   // logits padded to 7 col-tiles of 16
#define EPS 1e-5f

typedef __attribute__((ext_vector_type(16))) __bf16 v16bf;
typedef __attribute__((ext_vector_type(8)))  float  v8f;

// ---------------------------------------------------------------------------
// init (float4): zero scatter target (bufB), zero emb out, deg=1, stats=0
// ---------------------------------------------------------------------------
__global__ void k_init(float4* h1agg, float4* embout, float4* deg, float4* stats) {
    size_t i = (size_t)blockIdx.x * blockDim.x + threadIdx.x;
    const float4 z = make_float4(0.f, 0.f, 0.f, 0.f);
    const float4 o = make_float4(1.f, 1.f, 1.f, 1.f);   // self-loop degree
    if (i < (size_t)N_NODES * HID / 4) h1agg[i]  = z;
    if (i < (size_t)N_NODES * EMB / 4) embout[i] = z;
    if (i < N_NODES / 4)               deg[i]    = o;
    if (i < 128)                       stats[i]  = z;
}

// ---------------------------------------------------------------------------
// degree accumulation over dst, then deg -> deg^{-1/2} in place
// ---------------------------------------------------------------------------
__global__ void k_deg(const long long* dst, float* deg) {
    size_t e = (size_t)blockIdx.x * blockDim.x + threadIdx.x;
    if (e >= E_EDGES) return;
    long long d = dst[e];
    __hip_atomic_fetch_add(&deg[d], 1.0f, __ATOMIC_RELAXED, __HIP_MEMORY_SCOPE_AGENT);
}

__global__ void k_dis(float* deg) {
    size_t i = (size_t)blockIdx.x * blockDim.x + threadIdx.x;
    if (i < N_NODES) deg[i] = rsqrtf(deg[i]);   // deg >= 1 always (self loops)
}

// ---------------------------------------------------------------------------
// WMMA bf16 GEMM: C[M,NCOLS] = A[M,KD] @ B   (f32 in/out, bf16 operands)
//   BN_RELU:  apply per-K-channel scale/shift + ReLU to A while loading
//   BCENTERS: B[k][n] = centers[n*KD + k], zero for n >= KCLU
// one wave -> one 16x16 tile; 8 waves / block (wave32)
// ---------------------------------------------------------------------------
template<int KD, int NCOLS, int BN_RELU, int BCENTERS>
__global__ void k_gemm(const float* __restrict__ A, const float* __restrict__ B,
                       float* __restrict__ C,
                       const float* __restrict__ scale, const float* __restrict__ shift) {
    const int lane = threadIdx.x & 31;
    const int wv   = threadIdx.x >> 5;
    const int tiles_n = NCOLS / 16;
    const long total  = (long)(N_NODES / 16) * tiles_n;
    long tile = (long)blockIdx.x * 8 + wv;
    if (tile >= total) return;                 // wave-uniform: EXEC stays all-1s
    const int rt = (int)(tile / tiles_n);
    const int ct = (int)(tile % tiles_n);
    const int row0 = rt * 16, col0 = ct * 16;
    const int m = lane & 15, h = lane >> 4;

    v8f acc = {};
    #pragma unroll
    for (int k0 = 0; k0 < KD; k0 += 32) {
        // ---- A fragment (lanes 0-15: K = k0..k0+15, lanes 16-31: +16) ----
        v16bf a;
        const float* ap = A + (size_t)(row0 + m) * KD + k0;
        #pragma unroll
        for (int j = 0; j < 16; ++j) {
            int kk = h * 16 + j;
            float v = ap[kk];
            if (BN_RELU) {
                int ch = k0 + kk;
                v = fmaxf(v * scale[ch] + shift[ch], 0.0f);
            }
            a[j] = (__bf16)v;
        }
        // ---- B fragment (lanes 0-15: K = k0..k0+15, lanes 16-31: +16) ----
        v16bf b;
        #pragma unroll
        for (int j = 0; j < 16; ++j) {
            int kk  = k0 + h * 16 + j;
            int col = col0 + m;
            float v;
            if (BCENTERS) v = (col < KCLU) ? B[(size_t)col * KD + kk] : 0.0f;
            else          v = B[(size_t)kk * NCOLS + col];
            b[j] = (__bf16)v;
        }
        acc = __builtin_amdgcn_wmma_f32_16x16x32_bf16(
                  false, a, false, b, (short)0, acc, false, false);
    }
    // ---- D store: VGPR r -> M = h*8 + r, N = lane&15 ----
    #pragma unroll
    for (int r = 0; r < 8; ++r)
        C[((size_t)row0 + h * 8 + r) * NCOLS + col0 + m] = acc[r];
}

// ---------------------------------------------------------------------------
// edge scatter: out[dst] += in[src] * dis[src]*dis[dst], self loops appended
// ONE WAVE PER EDGE: e is wave-uniform -> edge indices / degrees become
// scalar (s_load) traffic; each lane handles F/32 contiguous floats so the
// gather coalesces into global_load_b128/b64; atomics stay native f32.
// ---------------------------------------------------------------------------
template<int F>
__global__ void k_scatter(const long long* __restrict__ src, const long long* __restrict__ dst,
                          const float* __restrict__ dis,
                          const float* __restrict__ in, float* __restrict__ out) {
    constexpr int VEC = F / 32;            // floats per lane (4 or 2)
    const int lane = threadIdx.x & 31;
    long e = (long)blockIdx.x * 8 + (threadIdx.x >> 5);   // 8 waves/block
    const long total = (long)E_EDGES + N_NODES;
    if (e >= total) return;                // wave-uniform exit
    long s, d;
    if (e < E_EDGES) { s = (long)src[e]; d = (long)dst[e]; }
    else             { s = d = e - E_EDGES; }
    float nrm = dis[s] * dis[d];
    const float* ip = in  + (size_t)s * F + lane * VEC;
    float*       op = out + (size_t)d * F + lane * VEC;
    float v[VEC];
    #pragma unroll
    for (int j = 0; j < VEC; ++j) v[j] = ip[j];          // merges to b128/b64
    #pragma unroll
    for (int j = 0; j < VEC; ++j)
        __hip_atomic_fetch_add(&op[j], v[j] * nrm,
                               __ATOMIC_RELAXED, __HIP_MEMORY_SCOPE_AGENT);
}

// ---------------------------------------------------------------------------
// BatchNorm training stats (b1 cancels inside BN, so it is skipped)
// ---------------------------------------------------------------------------
__global__ void k_bnstats(const float* __restrict__ H, float* sum, float* sumsq) {
    const int c = threadIdx.x & 127;
    const int rb = threadIdx.x >> 7;               // 0..1
    float s = 0.0f, q = 0.0f;
    for (int r = blockIdx.x * 2 + rb; r < N_NODES; r += gridDim.x * 2) {
        float v = H[(size_t)r * HID + c];
        s += v; q += v * v;
    }
    __hip_atomic_fetch_add(&sum[c],   s, __ATOMIC_RELAXED, __HIP_MEMORY_SCOPE_AGENT);
    __hip_atomic_fetch_add(&sumsq[c], q, __ATOMIC_RELAXED, __HIP_MEMORY_SCOPE_AGENT);
}

__global__ void k_bnfinish(float* stats, const float* gamma, const float* beta) {
    int c = threadIdx.x;                           // 128 threads
    float mu  = stats[c] * (1.0f / N_NODES);
    float var = stats[128 + c] * (1.0f / N_NODES) - mu * mu;
    float inv = rsqrtf(var + EPS);
    float sc  = gamma[c] * inv;
    stats[256 + c] = sc;                           // scale
    stats[384 + c] = beta[c] - mu * sc;            // shift
}

// ---------------------------------------------------------------------------
// emb += b2 (broadcast over 64 channels), float4
// ---------------------------------------------------------------------------
__global__ void k_addb2(float4* emb, const float4* b2) {
    size_t i = (size_t)blockIdx.x * blockDim.x + threadIdx.x;
    if (i >= (size_t)N_NODES * EMB / 4) return;
    float4 v = emb[i];
    float4 c = b2[i & (EMB / 4 - 1)];
    v.x += c.x; v.y += c.y; v.z += c.z; v.w += c.w;
    emb[i] = v;
}

// ---------------------------------------------------------------------------
// softmax over KCLU=100 logits (row stride LPAD), with temperature
// ---------------------------------------------------------------------------
__global__ void k_softmax(const float* __restrict__ logits, const float* __restrict__ temp,
                          float* __restrict__ soft) {
    __shared__ float red[128];
    const int node = blockIdx.x;
    const int t = threadIdx.x;
    const float invT = 1.0f / temp[0];
    float v = (t < KCLU) ? logits[(size_t)node * LPAD + t] * invT : -3.0e38f;
    red[t] = v;
    __syncthreads();
    for (int s = 64; s > 0; s >>= 1) {
        if (t < s) red[t] = fmaxf(red[t], red[t + s]);
        __syncthreads();
    }
    float mx = red[0];
    __syncthreads();
    float ex = (t < KCLU) ? __expf(v - mx) : 0.0f;
    red[t] = ex;
    __syncthreads();
    for (int s = 64; s > 0; s >>= 1) {
        if (t < s) red[t] += red[t + s];
        __syncthreads();
    }
    float inv = 1.0f / red[0];
    if (t < KCLU) soft[(size_t)node * KCLU + t] = ex * inv;
}

// ---------------------------------------------------------------------------
extern "C" void kernel_launch(void* const* d_in, const int* in_sizes, int n_in,
                              void* d_out, int out_size, void* d_ws, size_t ws_size,
                              hipStream_t stream) {
    const float*     x       = (const float*)d_in[0];
    const long long* ei      = (const long long*)d_in[1];   // [2,E] int64
    const float*     W1      = (const float*)d_in[2];
    // d_in[3] = b1 : cancels exactly inside BatchNorm -> unused
    const float*     gamma   = (const float*)d_in[4];
    const float*     beta    = (const float*)d_in[5];
    const float*     W2      = (const float*)d_in[6];
    const float*     b2      = (const float*)d_in[7];
    const float*     centers = (const float*)d_in[8];
    const float*     temp    = (const float*)d_in[9];

    float* out  = (float*)d_out;
    float* emb  = out;                                    // [N,64]
    float* soft = out + (size_t)N_NODES * EMB;            // [N,100]

    float* ws    = (float*)d_ws;
    float* deg   = ws;                                    // N (becomes dis)
    float* stats = ws + N_NODES;                          // 512
    float* bufA  = stats + 512;                           // N*128
    float* bufB  = bufA + (size_t)N_NODES * HID;          // N*128

    const long long* srcp = ei;
    const long long* dstp = ei + E_EDGES;

    // 1. init (float4 stores; covers the largest zeroed region N*128/4)
    {
        size_t tot = (size_t)N_NODES * HID / 4;
        k_init<<<(unsigned)((tot + 255) / 256), 256, 0, stream>>>(
            (float4*)bufB, (float4*)emb, (float4*)deg, (float4*)stats);
    }
    // 2. degrees + dis
    k_deg<<<(E_EDGES + 255) / 256, 256, 0, stream>>>(dstp, deg);
    k_dis<<<(N_NODES + 255) / 256, 256, 0, stream>>>(deg);

    // 3. XW1 = x @ W1   (bufA)
    k_gemm<IN_DIM, HID, 0, 0><<<(N_NODES / 16) * (HID / 16) / 8, 256, 0, stream>>>(
        x, W1, bufA, nullptr, nullptr);

    // 4. scatter1: bufB[dst] += XW1[src]*norm  (1 wave/edge, 8 edges/block)
    {
        long blocks = ((long)E_EDGES + N_NODES + 7) / 8;
        k_scatter<HID><<<(unsigned)blocks, 256, 0, stream>>>(srcp, dstp, deg, bufA, bufB);
    }

    // 5. BN stats + fold gamma/beta into scale/shift
    k_bnstats<<<1024, 256, 0, stream>>>(bufB, stats, stats + 128);
    k_bnfinish<<<1, 128, 0, stream>>>(stats, gamma, beta);

    // 6. HW2 = relu(BN(bufB)) @ W2  (BN+ReLU fused into A load) -> bufA
    k_gemm<HID, EMB, 1, 0><<<(N_NODES / 16) * (EMB / 16) / 8, 256, 0, stream>>>(
        bufB, W2, bufA, stats + 256, stats + 384);

    // 7. scatter2: emb[dst] += HW2[src]*norm  (1 wave/edge)
    {
        long blocks = ((long)E_EDGES + N_NODES + 7) / 8;
        k_scatter<EMB><<<(unsigned)blocks, 256, 0, stream>>>(srcp, dstp, deg, bufA, emb);
    }

    // 8. emb += b2
    k_addb2<<<(unsigned)(((size_t)N_NODES * EMB / 4 + 255) / 256), 256, 0, stream>>>(
        (float4*)emb, (const float4*)b2);

    // 9. logits = emb @ centers^T (padded to 112 cols) -> bufB
    {
        long tiles = (long)(N_NODES / 16) * (LPAD / 16);
        k_gemm<EMB, LPAD, 0, 1><<<(unsigned)((tiles + 7) / 8), 256, 0, stream>>>(
            emb, centers, bufB, nullptr, nullptr);
    }

    // 10. softmax with temperature -> soft
    k_softmax<<<N_NODES, 128, 0, stream>>>(bufB, temp, soft);
}
